// QORBlock_90177133347669
// MI455X (gfx1250) — compile-verified, matrix operand order checked by reading, not run on hardware
//
#include <hip/hip_runtime.h>
#include <cstdint>
#include <cstddef>

// ---------------------------------------------------------------------------
// Types / constants
// ---------------------------------------------------------------------------
typedef __attribute__((ext_vector_type(16))) __bf16 v16bf;
typedef __attribute__((ext_vector_type(8)))  float  v8f;
typedef __attribute__((ext_vector_type(4)))  unsigned int v4u;
typedef __attribute__((ext_vector_type(8)))  int    v8i_t;
typedef __attribute__((ext_vector_type(4)))  int    v4i_t;
typedef unsigned short u16;
typedef unsigned int   u32;
typedef unsigned long long u64;

constexpr int Bc  = 2;
constexpr int Tc  = 1024;
constexpr int Cc  = 2048;
constexpr int Hc  = 16;
constexpr int Dhc = 128;
constexpr int Mrows = Bc * Tc;      // 2048 token rows
constexpr float EPSc = 1e-6f;

#define DEVINL __device__ __forceinline__

#if defined(__has_builtin)
#  if __has_builtin(__builtin_amdgcn_tensor_load_to_lds)
#    define HAVE_TDM 1
#  else
#    define HAVE_TDM 0
#  endif
#else
#  define HAVE_TDM 0
#endif

union Frag {
  v16bf v;
  uint4 q[2];
};

DEVINL u16 f2bf(float f) {
  u32 u = __builtin_bit_cast(u32, f);
  u32 r = u + 0x7FFFu + ((u >> 16) & 1u);   // round-to-nearest-even
  return (u16)(r >> 16);
}

DEVINL v8f wmma_bf16(const Frag &a, const Frag &b, v8f c) {
  // D = A(16x32 bf16) * B(32x16 bf16) + C(16x16 f32)
  return __builtin_amdgcn_wmma_f32_16x16x32_bf16(
      /*neg_a=*/false, a.v, /*neg_b=*/false, b.v,
      /*c_mod=*/(short)0, c, /*reuse_a=*/false, /*reuse_b=*/false);
}

#if HAVE_TDM
// ---------------------------------------------------------------------------
// Tensor Data Mover: DMA one 128-row x 32-col bf16 tile (row stride K elems)
// from global memory into LDS. Hardware padding: 4 DWORDs (16B) inserted
// after every 16 DWORDs (64B row) -> LDS row stride 80B, matching the
// bank-conflict-free fragment-read layout. Tracked with TENSORcnt.
//   D# group0: [1:0]=count=1, [63:32]=lds_addr, [120:64]=global addr,
//              [127:126]=type=2
//   D# group1: [17:16]=data_size(1=2B), [20]=pad_en, [24:22]=pad_interval(3:16DW),
//              [31:25]=pad_amount(3:4DW), [79:48]=tensor_dim0=K,
//              [111:80]=tensor_dim1=rows_total, [127:112]=tile_dim0=32,
//              [143:128]=tile_dim1=128, [207:160]=tensor_dim0_stride=K
// ---------------------------------------------------------------------------
DEVINL void tdm_load_tile(const u16* g, u32 lds_off, int K, int rows) {
  u64 ga = (u64)(size_t)g;
  v4u g0;
  g0[0] = 1u;                                   // count=1, valid user D#
  g0[1] = lds_off;                              // LDS byte address
  g0[2] = (u32)ga;                              // global addr [31:0]
  g0[3] = (u32)((ga >> 32) & 0x01FFFFFFu) | 0x80000000u;  // [56:32] | type=2
  v8i_t g1;
  g1[0] = (int)((1u << 16) | (1u << 20) | (3u << 22) | (3u << 25));
  g1[1] = (int)(((u32)K & 0xFFFFu) << 16);                       // tdim0 lo
  g1[2] = (int)((((u32)K >> 16) & 0xFFFFu) |
                (((u32)rows & 0xFFFFu) << 16));                  // tdim0 hi|tdim1 lo
  g1[3] = (int)((((u32)rows >> 16) & 0xFFFFu) | (32u << 16));    // tdim1 hi|tile0=32
  g1[4] = 128;                                                   // tile_dim1=128
  g1[5] = K;                                                     // dim0 stride lo
  g1[6] = 0;
  g1[7] = 0;
  v4i_t z = {0, 0, 0, 0};
#if __clang_major__ >= 23
  v8i_t z8 = {0, 0, 0, 0, 0, 0, 0, 0};
  __builtin_amdgcn_tensor_load_to_lds(g0, g1, z, z, z8, 0);
#else
  __builtin_amdgcn_tensor_load_to_lds(g0, g1, z, z, 0);
#endif
}
#endif  // HAVE_TDM

// ---------------------------------------------------------------------------
// fp32 -> bf16 conversion (optionally fused a+b, used to fold sm_W + sm_dW)
// ---------------------------------------------------------------------------
__global__ __launch_bounds__(256)
void cvt_bf16_kernel(const float* __restrict__ a, const float* __restrict__ b,
                     u16* __restrict__ o, int n) {
  int i = blockIdx.x * 256 + threadIdx.x;
  int stride = gridDim.x * 256;
  for (; i < n; i += stride) {
    float v = a[i];
    if (b) v += b[i];
    o[i] = f2bf(v);
  }
}

// ---------------------------------------------------------------------------
// RMSNorm: one block per row of length C=2048, bf16 output
// ---------------------------------------------------------------------------
__global__ __launch_bounds__(256)
void rmsnorm_bf16_kernel(const float* __restrict__ x, const float* __restrict__ w,
                         u16* __restrict__ o) {
  __shared__ float red[256];
  const int row = blockIdx.x;
  const float* xr = x + (size_t)row * Cc;
  float loc[Cc / 256];
  float ss = 0.f;
#pragma unroll
  for (int j = 0; j < Cc / 256; ++j) {
    float v = xr[threadIdx.x + j * 256];
    loc[j] = v;
    ss += v * v;
  }
  red[threadIdx.x] = ss;
  __syncthreads();
  for (int off = 128; off > 0; off >>= 1) {
    if (threadIdx.x < off) red[threadIdx.x] += red[threadIdx.x + off];
    __syncthreads();
  }
  const float rstd = rsqrtf(red[0] * (1.0f / Cc) + EPSc);
  u16* orow = o + (size_t)row * Cc;
#pragma unroll
  for (int j = 0; j < Cc / 256; ++j) {
    int idx = threadIdx.x + j * 256;
    orow[idx] = f2bf(loc[j] * rstd * w[idx]);
  }
}

// ---------------------------------------------------------------------------
// Tiled WMMA GEMM:  out(MxN) = A(MxK,bf16) * W(NxK,bf16)^T  [+ epilogue]
//   block tile 128x128, 8 waves (2x4), wave tile 64x32, K-step 32
//   Double-buffered LDS fed by the Tensor Data Mover (wave 0 issues the
//   tile descriptors; s_wait_tensorcnt + barrier publishes the buffer).
// ---------------------------------------------------------------------------
enum { EPI_F32 = 0, EPI_RES = 1, EPI_SILU = 2, EPI_VT = 3 };

template <int EPI>
__global__ __launch_bounds__(256)
void gemm_bf16_kernel(const u16* __restrict__ A, const u16* __restrict__ W,
                      void* __restrict__ outp, const float* __restrict__ res,
                      const float* __restrict__ bias, int N, int K, int Mtot) {
  __shared__ u16 ldsA[2][128 * 40];   // 32 bf16 cols padded to 40 (80B rows)
  __shared__ u16 ldsB[2][128 * 40];

  const int tid   = threadIdx.x;
  const int lane  = tid & 31;
  const int wave  = tid >> 5;
  const int waveM = wave >> 2;      // 0..1
  const int waveN = wave & 3;       // 0..3
  const int blockM = blockIdx.y * 128;
  const int blockN = blockIdx.x * 128;

  const int nl   = lane & 15;
  const int koff = (lane < 16) ? 0 : 8;   // K half-select per ISA frag layout
  const int hi8  = (lane >> 4) << 3;

  v8f acc[4][2];
#pragma unroll
  for (int mt = 0; mt < 4; ++mt)
#pragma unroll
    for (int nt = 0; nt < 2; ++nt)
#pragma unroll
      for (int r = 0; r < 8; ++r) acc[mt][nt][r] = 0.f;

  const int aRowBase = waveM * 64 + nl;
  const int bRowBase = waveN * 32 + nl;

#if HAVE_TDM
  const u32 aoff[2] = {(u32)(size_t)(void*)&ldsA[0][0],
                       (u32)(size_t)(void*)&ldsA[1][0]};
  const u32 boff[2] = {(u32)(size_t)(void*)&ldsB[0][0],
                       (u32)(size_t)(void*)&ldsB[1][0]};
  if (wave == 0) {
    tdm_load_tile(A + (size_t)blockM * K, aoff[0], K, Mtot);
    tdm_load_tile(W + (size_t)blockN * K, boff[0], K, N);
  }
#endif

  int nb = 0;
  for (int k0 = 0; k0 < K; k0 += 32, nb ^= 1) {
#if HAVE_TDM
    if (wave == 0) {
      if (k0 + 32 < K) {
        // prefetch next tile pair into the other buffer, then wait for the
        // current pair (tensor ops from one wave complete in order)
        tdm_load_tile(A + (size_t)blockM * K + (k0 + 32), aoff[nb ^ 1], K, Mtot);
        tdm_load_tile(W + (size_t)blockN * K + (k0 + 32), boff[nb ^ 1], K, N);
        __builtin_amdgcn_s_wait_tensorcnt(2);
      } else {
        __builtin_amdgcn_s_wait_tensorcnt(0);
      }
    }
    __syncthreads();
#else
    // cooperative global -> LDS fallback (each thread: 2 x 16B per matrix)
#pragma unroll
    for (int i = 0; i < 2; ++i) {
      int s   = tid * 2 + i;       // 0..511
      int row = s >> 2;            // 0..127
      int col = (s & 3) << 3;      // 0,8,16,24
      const u16* ga = A + (size_t)(blockM + row) * K + k0 + col;
      const u16* gw = W + (size_t)(blockN + row) * K + k0 + col;
      *(uint4*)&ldsA[nb][row * 40 + col] = *(const uint4*)ga;
      *(uint4*)&ldsB[nb][row * 40 + col] = *(const uint4*)gw;
      if (k0 + 32 < K) {
        __builtin_prefetch(ga + 32, 0, 1);
        __builtin_prefetch(gw + 32, 0, 1);
      }
    }
    __syncthreads();
#endif

    Frag af[4], bfr[2];
#pragma unroll
    for (int mt = 0; mt < 4; ++mt) {
      const u16* p = &ldsA[nb][(aRowBase + mt * 16) * 40 + koff];
      af[mt].q[0] = *(const uint4*)p;
      af[mt].q[1] = *(const uint4*)(p + 16);
    }
#pragma unroll
    for (int nt = 0; nt < 2; ++nt) {
      const u16* p = &ldsB[nb][(bRowBase + nt * 16) * 40 + koff];
      bfr[nt].q[0] = *(const uint4*)p;
      bfr[nt].q[1] = *(const uint4*)(p + 16);
    }
#pragma unroll
    for (int mt = 0; mt < 4; ++mt)
#pragma unroll
      for (int nt = 0; nt < 2; ++nt)
        acc[mt][nt] = wmma_bf16(af[mt], bfr[nt], acc[mt][nt]);
    __syncthreads();
  }

  // epilogue (C layout: lanes 0-15 -> M=r, lanes 16-31 -> M=r+8; N = lane&15)
#pragma unroll
  for (int mt = 0; mt < 4; ++mt)
#pragma unroll
    for (int nt = 0; nt < 2; ++nt)
#pragma unroll
      for (int r = 0; r < 8; ++r) {
        int m = blockM + waveM * 64 + mt * 16 + r + hi8;
        int n = blockN + waveN * 32 + nt * 16 + nl;
        float v = acc[mt][nt][r];
        size_t oi = (size_t)m * N + n;
        if (EPI == EPI_F32) {
          ((float*)outp)[oi] = v;
        } else if (EPI == EPI_RES) {
          if (bias) v += bias[n];
          ((float*)outp)[oi] = res[oi] + v;
        } else if (EPI == EPI_SILU) {
          v += bias[n];
          float sv = v / (1.0f + __expf(-v));
          ((u16*)outp)[oi] = f2bf(sv);
        } else {  // EPI_VT: write v transposed per head: (b,h,d,t) bf16
          int b = m / Tc, t = m % Tc;
          int hh = n >> 7, d = n & 127;
          ((u16*)outp)[(((size_t)(b * Hc + hh)) * Dhc + d) * Tc + t] = f2bf(v);
        }
      }
}

// ---------------------------------------------------------------------------
// RoPE: q,k fp32 (B*T, C) -> bf16 (B,H,T,Dh); folds 1/sqrt(Dh) into q
// ---------------------------------------------------------------------------
__global__ __launch_bounds__(256)
void rope_kernel(const float* __restrict__ qf, const float* __restrict__ kf,
                 const float* __restrict__ cs, const float* __restrict__ sn,
                 u16* __restrict__ qo, u16* __restrict__ ko) {
  int idx = blockIdx.x * 256 + threadIdx.x;  // B*T*H*64 = 2M threads
  int d = idx & 63;
  int h = (idx >> 6) & 15;
  int t = (idx >> 10) & 1023;
  int b = idx >> 20;
  size_t ib = ((size_t)(b * Tc + t)) * Cc + h * Dhc;
  float c = cs[t * 64 + d], s = sn[t * 64 + d];
  float q1 = qf[ib + d], q2 = qf[ib + d + 64];
  float k1 = kf[ib + d], k2 = kf[ib + d + 64];
  const float scale = 0.088388347648318447f;  // 1/sqrt(128)
  size_t ob = (((size_t)(b * Hc + h)) * Tc + t) * Dhc;
  qo[ob + d]      = f2bf((q1 * c - q2 * s) * scale);
  qo[ob + d + 64] = f2bf((q2 * c + q1 * s) * scale);
  ko[ob + d]      = f2bf(k1 * c - k2 * s);
  ko[ob + d + 64] = f2bf(k2 * c + k1 * s);
}

// ---------------------------------------------------------------------------
// Causal flash attention, all-WMMA.
//   grid = (T/128, B*H), 8 waves/block, each wave: one 16-row q tile.
//   kv streamed in 32-wide chunks; online softmax; P transposed via LDS.
// ---------------------------------------------------------------------------
__global__ __launch_bounds__(256)
void flash_attn_kernel(const u16* __restrict__ qb, const u16* __restrict__ kb,
                       const u16* __restrict__ vtb, u16* __restrict__ ob) {
  __shared__ u16 pbuf[8][16 * 40];  // per-wave P scratch (16x32, padded rows)

  const int tid  = threadIdx.x;
  const int lane = tid & 31;
  const int wave = tid >> 5;
  const int bh   = blockIdx.y;
  const int b    = bh >> 4;       // / H
  const int h    = bh & 15;
  const int qbase = blockIdx.x * 128 + wave * 16;
  const int nl   = lane & 15;
  const int koff = (lane < 16) ? 0 : 8;
  const int hi8  = (lane >> 4) << 3;

  // q tile as 4 A-fragments (Dh = 128 = 4 * K32)
  Frag aq[4];
  {
    const u16* qp = qb + ((size_t)bh * Tc + qbase + nl) * Dhc;
#pragma unroll
    for (int c = 0; c < 4; ++c) {
      const u16* p = qp + c * 32 + koff;
      aq[c].q[0] = *(const uint4*)p;
      aq[c].q[1] = *(const uint4*)(p + 16);
    }
  }

  v8f accO[8];
  float mrun[8], lrun[8];
#pragma unroll
  for (int dt = 0; dt < 8; ++dt)
#pragma unroll
    for (int r = 0; r < 8; ++r) accO[dt][r] = 0.f;
#pragma unroll
  for (int r = 0; r < 8; ++r) { mrun[r] = -3.0e38f; lrun[r] = 0.f; }

  u16* pb = &pbuf[wave][0];

  for (int kv0 = 0; kv0 < qbase + 16; kv0 += 32) {
    // ---- scores: two 16x16 tiles (kv0..+15, kv0+16..+31) ----
    v8f s0, s1;
#pragma unroll
    for (int r = 0; r < 8; ++r) { s0[r] = 0.f; s1[r] = 0.f; }
#pragma unroll
    for (int c = 0; c < 4; ++c) {
      Frag bk0, bk1;
      const u16* kp = kb + ((size_t)bh * Tc + kv0 + nl) * Dhc + c * 32 + koff;
      bk0.q[0] = *(const uint4*)kp;
      bk0.q[1] = *(const uint4*)(kp + 16);
      const u16* kp1 = kp + 16 * Dhc;
      bk1.q[0] = *(const uint4*)kp1;
      bk1.q[1] = *(const uint4*)(kp1 + 16);
      s0 = wmma_bf16(aq[c], bk0, s0);
      s1 = wmma_bf16(aq[c], bk1, s1);
    }

    // ---- online softmax (row reductions across 16-lane halves) ----
    float alpha[8];
#pragma unroll
    for (int r = 0; r < 8; ++r) {
      const int qrow = qbase + r + hi8;
      float v0 = (kv0 + nl      <= qrow) ? s0[r] : -3.0e38f;
      float v1 = (kv0 + 16 + nl <= qrow) ? s1[r] : -3.0e38f;
      float mx = fmaxf(v0, v1);
#pragma unroll
      for (int off = 8; off >= 1; off >>= 1)
        mx = fmaxf(mx, __shfl_xor(mx, off, 16));
      float mnew = fmaxf(mrun[r], mx);
      alpha[r] = __expf(mrun[r] - mnew);
      mrun[r] = mnew;
      float p0 = __expf(v0 - mnew);
      float p1 = __expf(v1 - mnew);
      float rs = p0 + p1;
#pragma unroll
      for (int off = 8; off >= 1; off >>= 1)
        rs += __shfl_xor(rs, off, 16);
      lrun[r] = lrun[r] * alpha[r] + rs;
      // stash P into per-wave LDS (C-layout -> row-major 16x32)
      pb[(r + hi8) * 40 + nl]      = f2bf(p0);
      pb[(r + hi8) * 40 + 16 + nl] = f2bf(p1);
    }
#pragma unroll
    for (int dt = 0; dt < 8; ++dt)
#pragma unroll
      for (int r = 0; r < 8; ++r) accO[dt][r] *= alpha[r];

    // ---- P as A-fragment (transpose via LDS; wave-private, DScnt ordered) ----
    Frag ap;
    {
      const u16* pr = pb + nl * 40 + koff;
      ap.q[0] = *(const uint4*)pr;
      ap.q[1] = *(const uint4*)(pr + 16);
    }
    // ---- O += P * V (v stored transposed: (b,h,d,t)) ----
#pragma unroll
    for (int dt = 0; dt < 8; ++dt) {
      Frag bv;
      const u16* vp = vtb + ((size_t)bh * Dhc + dt * 16 + nl) * Tc + kv0 + koff;
      bv.q[0] = *(const uint4*)vp;
      bv.q[1] = *(const uint4*)(vp + 16);
      accO[dt] = wmma_bf16(ap, bv, accO[dt]);
    }
  }

#pragma unroll
  for (int r = 0; r < 8; ++r) lrun[r] = 1.0f / lrun[r];
#pragma unroll
  for (int dt = 0; dt < 8; ++dt)
#pragma unroll
    for (int r = 0; r < 8; ++r) {
      int trow = qbase + r + hi8;
      size_t oi = ((size_t)(b * Tc + trow)) * Cc + h * Dhc + dt * 16 + nl;
      ob[oi] = f2bf(accO[dt][r] * lrun[r]);
    }
}

// ---------------------------------------------------------------------------
// Host orchestration
// ---------------------------------------------------------------------------
extern "C" void kernel_launch(void* const* d_in, const int* in_sizes, int n_in,
                              void* d_out, int out_size, void* d_ws, size_t ws_size,
                              hipStream_t stream) {
  (void)in_sizes; (void)n_in; (void)out_size; (void)ws_size;

  const float* x           = (const float*)d_in[0];
  const float* wq          = (const float*)d_in[1];
  const float* wk          = (const float*)d_in[2];
  const float* wv          = (const float*)d_in[3];
  const float* wo          = (const float*)d_in[4];
  const float* attn_norm_w = (const float*)d_in[5];
  const float* sm_norm_w   = (const float*)d_in[6];
  const float* sm_W        = (const float*)d_in[7];
  const float* sm_dW       = (const float*)d_in[8];
  const float* c_norm[3] = {(const float*)d_in[9],  (const float*)d_in[14], (const float*)d_in[19]};
  const float* c_w1[3]   = {(const float*)d_in[10], (const float*)d_in[15], (const float*)d_in[20]};
  const float* c_b1[3]   = {(const float*)d_in[11], (const float*)d_in[16], (const float*)d_in[21]};
  const float* c_w2[3]   = {(const float*)d_in[12], (const float*)d_in[17], (const float*)d_in[22]};
  const float* c_b2[3]   = {(const float*)d_in[13], (const float*)d_in[18], (const float*)d_in[23]};
  const float* rope_cos  = (const float*)d_in[24];
  const float* rope_sin  = (const float*)d_in[25];
  const int FF[3] = {4096, 2048, 2048};

  // bump allocator on workspace
  char* ws = (char*)d_ws;
  size_t off = 0;
  auto alloc = [&](size_t bytes) -> void* {
    void* p = (void*)(ws + off);
    off += (bytes + 255) & ~(size_t)255;
    return p;
  };

  const size_t CCn = (size_t)Cc * Cc;  // 4.19M elements
  u16* wq_b  = (u16*)alloc(CCn * 2);
  u16* wk_b  = (u16*)alloc(CCn * 2);
  u16* wv_b  = (u16*)alloc(CCn * 2);
  u16* wo_b  = (u16*)alloc(CCn * 2);
  u16* smw_b = (u16*)alloc(CCn * 2);
  u16* w1_b[3], *w2_b[3];
  for (int l = 0; l < 3; ++l) {
    w1_b[l] = (u16*)alloc((size_t)FF[l] * Cc * 2);
    w2_b[l] = (u16*)alloc((size_t)FF[l] * Cc * 2);
  }
  u16*   nrm_b = (u16*)alloc((size_t)Mrows * Cc * 2);
  float* qf    = (float*)alloc((size_t)Mrows * Cc * 4);  // reused as act_b later
  float* kf    = (float*)alloc((size_t)Mrows * Cc * 4);  // reused as xcur later
  u16*   q_b   = (u16*)alloc((size_t)Mrows * Cc * 2);
  u16*   k_b   = (u16*)alloc((size_t)Mrows * Cc * 2);
  u16*   vt_b  = (u16*)alloc((size_t)Mrows * Cc * 2);
  u16*   o_b   = (u16*)alloc((size_t)Mrows * Cc * 2);
  u16*   act_b = (u16*)qf;   // MLP hidden (<= M * 4096 bf16 = 16MB, fits)
  float* xcur  = kf;         // running residual (after RoPE consumed kf)

  const dim3 blk(256);
  auto cvt = [&](const float* a, const float* b, u16* o, size_t n) {
    cvt_bf16_kernel<<<dim3(512), blk, 0, stream>>>(a, b, o, (int)n);
  };

  // --- weight conversion to bf16 (sm_dW folded into sm_W) ---
  cvt(wq, nullptr, wq_b, CCn);
  cvt(wk, nullptr, wk_b, CCn);
  cvt(wv, nullptr, wv_b, CCn);
  cvt(wo, nullptr, wo_b, CCn);
  cvt(sm_W, sm_dW, smw_b, CCn);
  for (int l = 0; l < 3; ++l) {
    cvt(c_w1[l], nullptr, w1_b[l], (size_t)FF[l] * Cc);
    cvt(c_w2[l], nullptr, w2_b[l], (size_t)FF[l] * Cc);
  }

  const dim3 gCC(Cc / 128, Mrows / 128);  // (16,16)

  // --- attention ---
  rmsnorm_bf16_kernel<<<Mrows, blk, 0, stream>>>(x, attn_norm_w, nrm_b);
  gemm_bf16_kernel<EPI_F32><<<gCC, blk, 0, stream>>>(nrm_b, wq_b, qf, nullptr, nullptr, Cc, Cc, Mrows);
  gemm_bf16_kernel<EPI_F32><<<gCC, blk, 0, stream>>>(nrm_b, wk_b, kf, nullptr, nullptr, Cc, Cc, Mrows);
  gemm_bf16_kernel<EPI_VT ><<<gCC, blk, 0, stream>>>(nrm_b, wv_b, vt_b, nullptr, nullptr, Cc, Cc, Mrows);
  rope_kernel<<<(Bc * Tc * Hc * 64) / 256, blk, 0, stream>>>(qf, kf, rope_cos, rope_sin, q_b, k_b);
  flash_attn_kernel<<<dim3(Tc / 128, Bc * Hc), blk, 0, stream>>>(q_b, k_b, vt_b, o_b);
  gemm_bf16_kernel<EPI_RES><<<gCC, blk, 0, stream>>>(o_b, wo_b, xcur, x, nullptr, Cc, Cc, Mrows);

  // --- shared "slow memory" projection: x += nm @ (sm_W + sm_dW)^T ---
  rmsnorm_bf16_kernel<<<Mrows, blk, 0, stream>>>(xcur, sm_norm_w, nrm_b);
  gemm_bf16_kernel<EPI_RES><<<gCC, blk, 0, stream>>>(nrm_b, smw_b, xcur, xcur, nullptr, Cc, Cc, Mrows);

  // --- MLP stack ---
  for (int l = 0; l < 3; ++l) {
    rmsnorm_bf16_kernel<<<Mrows, blk, 0, stream>>>(xcur, c_norm[l], nrm_b);
    dim3 g1(FF[l] / 128, Mrows / 128);
    gemm_bf16_kernel<EPI_SILU><<<g1, blk, 0, stream>>>(nrm_b, w1_b[l], act_b, nullptr, c_b1[l], FF[l], Cc, Mrows);
    float* outl = (l == 2) ? (float*)d_out : xcur;
    gemm_bf16_kernel<EPI_RES><<<gCC, blk, 0, stream>>>(act_b, w2_b[l], outl, xcur, c_b2[l], Cc, FF[l], Mrows);
  }
}
